// Block_55207509622872
// MI455X (gfx1250) — compile-verified, matrix operand order checked by reading, not run on hardware
//
#include <hip/hip_runtime.h>
#include <math.h>

// ---------------- problem constants (match reference) ----------------
static constexpr int Bc    = 2;
static constexpr int T     = 2048;
static constexpr int C     = 1024;
static constexpr int H     = 16;
static constexpr int EXP   = 8;
static constexpr int TOPK  = 2;
static constexpr int HD    = C / H;          // 64
static constexpr int N_TOK = Bc * T;         // 4096
static constexpr int CAP   = N_TOK * TOPK / EXP; // 1024
static constexpr int C4    = 4 * C;          // 4096
static constexpr float EPS = 1e-6f;

typedef __bf16 bf16_t;
typedef __attribute__((ext_vector_type(16))) __bf16 v16bf;
typedef __attribute__((ext_vector_type(8)))  float  v8f;
typedef __attribute__((ext_vector_type(4)))  unsigned int v4u;
typedef __attribute__((ext_vector_type(8)))  int v8i;
typedef __attribute__((ext_vector_type(4)))  int v4i;

#if defined(__has_builtin)
#if __has_builtin(__builtin_amdgcn_tensor_load_to_lds) && \
    __has_builtin(__builtin_amdgcn_s_wait_tensorcnt)
#define HAVE_TDM 1
#endif
#endif
#ifndef HAVE_TDM
#define HAVE_TDM 0
#endif

// ---------------- WMMA helpers (CDNA5: wave32, 16x16x32 bf16) ----------------
__device__ __forceinline__ v8f wmma_bf16(v16bf a, v16bf b, v8f c) {
  return __builtin_amdgcn_wmma_f32_16x16x32_bf16(false, a, false, b, (short)0, c,
                                                 false, false);
}

// Operand fragment in the 16x32 (rows x K) A-layout; B operands are supplied as
// B^T rows in the same layout (WMMA example loads A and B
// identically). Lane holds row = lane%16; lanes<16: K 0-7,16-23; lanes>=16:
// K 8-15,24-31. Both 8-half groups are contiguous -> two b128 loads.
// p = &M[row][kk] with contiguous K.
__device__ __forceinline__ v16bf frag_load_a(const bf16_t* p, int lane) {
  int kb = (lane & 16) ? 8 : 0;
  union { uint4 u[2]; v16bf v; } t;
  t.u[0] = *(const uint4*)(p + kb);
  t.u[1] = *(const uint4*)(p + kb + 16);
  return t.v;
}

// Block-cooperative core: 128 threads = 4 waves. All waves share one 64(N)x32(K)
// W tile staged in LDS per K-step (4x weight-traffic reuse, coalesced b128 loads);
// each wave owns a different 16-row M tile and produces a 16x64 f32 D tile.
// B fragments are read back from LDS as contiguous b128 (ds_load_b128) pairs.
__device__ __forceinline__ void gemm_core_blk(const bf16_t* __restrict__ A,
                                              const bf16_t* __restrict__ W, int K,
                                              int m0, int nblk, bf16_t* ldsw,
                                              v8f acc[4], int lane) {
  int ml = lane & 15;
  const bf16_t* arow = A + (size_t)(m0 + ml) * K;
  int sr = threadIdx.x >> 1;            // 0..63 : row of W tile
  int sc = (threadIdx.x & 1) * 16;      // 0 or 16 : col half of W tile
  const bf16_t* wrow = W + (size_t)(nblk + sr) * K + sc;
  for (int kk = 0; kk < K; kk += 32) {
    __syncthreads();                    // WAR vs previous iteration's reads
    *(uint4*)(ldsw + sr * 32 + sc) = *(const uint4*)(wrow + kk);
    __builtin_prefetch(wrow + kk + 512, 0, 0);   // global_prefetch_b8
    __syncthreads();                    // publish W tile
    v16bf af = frag_load_a(arow + kk, lane);
#pragma unroll
    for (int g = 0; g < 4; ++g) {
      // B^T rows (n-major) from LDS tile, pitch 32 halves -> 2x ds_load_b128
      v16bf bfrag = frag_load_a(ldsw + (size_t)(g * 16 + ml) * 32, lane);
      acc[g] = wmma_bf16(af, bfrag, acc[g]);
    }
  }
}

// ---------------- small utility kernels ----------------
__global__ void k_cvt_bf16(const float* __restrict__ s, bf16_t* __restrict__ d,
                           long n) { // n divisible by 4
  long i  = ((long)blockIdx.x * blockDim.x + threadIdx.x) * 4;
  long st = (long)gridDim.x * blockDim.x * 4;
  for (; i < n; i += st) {
    float4 v = *(const float4*)(s + i);
    union { bf16_t b[4]; uint2 u; } t;
    t.b[0] = (bf16_t)v.x; t.b[1] = (bf16_t)v.y;
    t.b[2] = (bf16_t)v.z; t.b[3] = (bf16_t)v.w;
    *(uint2*)(d + i) = t.u;
  }
}

__device__ float block_reduce_sum256(float v) {
  __shared__ float red[8];
  __shared__ float tot;
  int lane = threadIdx.x & 31, w = threadIdx.x >> 5;
#pragma unroll
  for (int m = 16; m >= 1; m >>= 1) v += __shfl_xor(v, m, 32);
  if (lane == 0) red[w] = v;
  __syncthreads();
  if (threadIdx.x == 0) {
    float s = 0.f;
    for (int i = 0; i < 8; ++i) s += red[i];
    tot = s;
  }
  __syncthreads();
  return tot;
}

// x = l0*x + l1*x0 ; a = rmsnorm(x) -> bf16 ; keep x (f32 residual)
__global__ void k_mix_rms(const float* __restrict__ x, const float* __restrict__ x0,
                          const float* __restrict__ lam, float* __restrict__ xres,
                          bf16_t* __restrict__ abf) {
  int row = blockIdx.x;
  float l0 = lam[0], l1 = lam[1];
  size_t base = (size_t)row * C;
  float v[4];
  float ss = 0.f;
#pragma unroll
  for (int i = 0; i < 4; ++i) {
    int c = threadIdx.x + i * 256;
    v[i]  = l0 * x[base + c] + l1 * x0[base + c];
    ss += v[i] * v[i];
  }
  float tot = block_reduce_sum256(ss);
  float inv = rsqrtf(tot / (float)C + EPS);
#pragma unroll
  for (int i = 0; i < 4; ++i) {
    int c = threadIdx.x + i * 256;
    xres[base + c] = v[i];
    abf[base + c]  = (bf16_t)(v[i] * inv);
  }
}

// h = rmsnorm(xin); produce f32 + bf16
__global__ void k_rms_h(const float* __restrict__ xin, float* __restrict__ hf,
                        bf16_t* __restrict__ hb) {
  int row = blockIdx.x;
  size_t base = (size_t)row * C;
  float v[4];
  float ss = 0.f;
#pragma unroll
  for (int i = 0; i < 4; ++i) {
    int c = threadIdx.x + i * 256;
    v[i]  = xin[base + c];
    ss += v[i] * v[i];
  }
  float tot = block_reduce_sum256(ss);
  float inv = rsqrtf(tot / (float)C + EPS);
#pragma unroll
  for (int i = 0; i < 4; ++i) {
    int c = threadIdx.x + i * 256;
    float hv = v[i] * inv;
    hf[base + c] = hv;
    hb[base + c] = (bf16_t)hv;
  }
}

// ---------------- QKV GEMM: qkv[z] = a @ W[z]^T  (W row-major [O][C]) ----------
__global__ void k_gemm_qkv(const bf16_t* __restrict__ A, const bf16_t* __restrict__ Wq,
                           float* __restrict__ out) {
  __shared__ bf16_t ldsw[64 * 32];
  int lane = threadIdx.x & 31, w = threadIdx.x >> 5;
  int nblk = blockIdx.x * 64;
  int m0   = blockIdx.y * 64 + w * 16;
  const bf16_t* W = Wq + (size_t)blockIdx.z * C * C;
  float* o = out + (size_t)blockIdx.z * N_TOK * C;
  v8f acc[4] = {};
  gemm_core_blk(A, W, C, m0, nblk, ldsw, acc, lane);
  int ml = lane & 15, hi = (lane >> 4) & 1;
#pragma unroll
  for (int g = 0; g < 4; ++g)
#pragma unroll
    for (int r = 0; r < 8; ++r)
      o[(size_t)(m0 + r + 8 * hi) * C + nblk + g * 16 + ml] = acc[g][r];
}

// per (b,h,t): rmsnorm q,k over HD, rotary; q,k -> [B][H][T][HD];
// v stored TRANSPOSED per head -> [B][H][HD][T] so P@V's B^T is row-major.
__global__ void k_qk_prep(const float* __restrict__ qkv, const float* __restrict__ lambp,
                          bf16_t* __restrict__ qb, bf16_t* __restrict__ kb,
                          bf16_t* __restrict__ vt) {
  int d = threadIdx.x;             // 0..63
  int t = blockIdx.x, hh = blockIdx.y, b = blockIdx.z;
  size_t base = ((size_t)b * T + t) * C + hh * HD + d;
  float q = qkv[base];
  float k = qkv[(size_t)N_TOK * C + base];
  float v = qkv[2 * (size_t)N_TOK * C + base];
  float lam = *lambp;
  v = (1.f - lam) * v + lam * v;
  __shared__ float sq[HD], sk[HD];
  sq[d] = q; sk[d] = k;
  __syncthreads();
  float ssq = 0.f, ssk = 0.f;
  for (int i = 0; i < HD; ++i) { ssq += sq[i] * sq[i]; ssk += sk[i] * sk[i]; }
  float rq = rsqrtf(ssq / (float)HD + EPS);
  float rk = rsqrtf(ssk / (float)HD + EPS);
  __syncthreads();
  sq[d] = q * rq; sk[d] = k * rk;
  __syncthreads();
  int j = d & 31;                          // rotary index into half-dim (32)
  float inv = (j < 16) ? powf(1024.f, -(float)j / 15.f) : 0.f;
  float th = (float)t * inv;
  float cs = cosf(th), sn = sinf(th);
  float qo, ko;
  if (d < 32) {
    qo = sq[d] * cs + sq[d + 32] * sn;
    ko = sk[d] * cs + sk[d + 32] * sn;
  } else {
    qo = -sq[d - 32] * sn + sq[d] * cs;
    ko = -sk[d - 32] * sn + sk[d] * cs;
  }
  size_t ob = (((size_t)b * H + hh) * T + t) * HD + d;
  qb[ob] = (bf16_t)qo;
  kb[ob] = (bf16_t)ko;
  vt[(((size_t)b * H + hh) * HD + d) * T + t] = (bf16_t)v;
}

// ---------------- flash attention: one wave per 16-query tile -------------------
__global__ void k_attn(const bf16_t* __restrict__ qb, const bf16_t* __restrict__ kbf,
                       const bf16_t* __restrict__ vtb, bf16_t* __restrict__ ob) {
  __shared__ bf16_t pl[16 * 32];
  int lane = threadIdx.x;
  int qt = blockIdx.x, hh = blockIdx.y, b = blockIdx.z;
  size_t hbase = (((size_t)b * H + hh) * T) * HD;
  const bf16_t* Q  = qb + hbase;
  const bf16_t* Kp = kbf + hbase;
  const bf16_t* VT = vtb + hbase;        // [HD][T] per head
  int ml = lane & 15, hi = (lane >> 4) & 1;
  v16bf qf0 = frag_load_a(Q + (size_t)(qt * 16 + ml) * HD + 0, lane);
  v16bf qf1 = frag_load_a(Q + (size_t)(qt * 16 + ml) * HD + 32, lane);
  v8f acc[4] = {};
  float mrow[8], lrow[8];
#pragma unroll
  for (int r = 0; r < 8; ++r) { mrow[r] = -INFINITY; lrow[r] = 0.f; }
  const int qhi = qt * 16 + 15;
  const float sc = 0.125f; // 1/sqrt(64)
  for (int kb0 = 0; kb0 <= qhi; kb0 += 32) {
    float s[2][8];
#pragma unroll
    for (int sub = 0; sub < 2; ++sub) {
      int kt = kb0 + sub * 16;
      if (kt <= qhi) {  // wave-uniform branch (EXEC stays all-ones for WMMA)
        v8f sa = {};
        // B^T = K rows (row-major) -> contiguous b128 fragment loads
        sa = wmma_bf16(qf0, frag_load_a(Kp + (size_t)(kt + ml) * HD + 0, lane), sa);
        sa = wmma_bf16(qf1, frag_load_a(Kp + (size_t)(kt + ml) * HD + 32, lane), sa);
        int kj = kt + ml;
#pragma unroll
        for (int r = 0; r < 8; ++r) {
          int qi = qt * 16 + r + 8 * hi;
          s[sub][r] = (kj <= qi) ? sa[r] * sc : -INFINITY;
        }
      } else {
#pragma unroll
        for (int r = 0; r < 8; ++r) s[sub][r] = -INFINITY;
      }
    }
    float alpha[8], pv0[8], pv1[8];
#pragma unroll
    for (int r = 0; r < 8; ++r) {
      float rm = fmaxf(s[0][r], s[1][r]);
      rm = fmaxf(rm, __shfl_xor(rm, 1, 32));
      rm = fmaxf(rm, __shfl_xor(rm, 2, 32));
      rm = fmaxf(rm, __shfl_xor(rm, 4, 32));
      rm = fmaxf(rm, __shfl_xor(rm, 8, 32));
      float mn  = fmaxf(mrow[r], rm);
      alpha[r]  = __expf(mrow[r] - mn);
      mrow[r]   = mn;
      pv0[r]    = __expf(s[0][r] - mn);
      pv1[r]    = __expf(s[1][r] - mn);
      float ps  = pv0[r] + pv1[r];
      ps += __shfl_xor(ps, 1, 32);
      ps += __shfl_xor(ps, 2, 32);
      ps += __shfl_xor(ps, 4, 32);
      ps += __shfl_xor(ps, 8, 32);
      lrow[r] = alpha[r] * lrow[r] + ps;
    }
#pragma unroll
    for (int g = 0; g < 4; ++g)
#pragma unroll
      for (int r = 0; r < 8; ++r) acc[g][r] *= alpha[r];
    __syncthreads(); // WAR vs previous iteration's LDS reads (single wave/block)
#pragma unroll
    for (int r = 0; r < 8; ++r) {
      int m = r + 8 * hi;
      pl[m * 32 + ml]      = (bf16_t)pv0[r];
      pl[m * 32 + 16 + ml] = (bf16_t)pv1[r];
    }
    __syncthreads(); // publish P tile before fragment reads
    v16bf pf = frag_load_a(pl + (size_t)ml * 32, lane);
#pragma unroll
    for (int g = 0; g < 4; ++g) {
      // B^T = V^T rows (d-major, pitch T) -> contiguous b128 fragment loads
      v16bf vf = frag_load_a(VT + (size_t)(g * 16 + ml) * T + kb0, lane);
      acc[g] = wmma_bf16(pf, vf, acc[g]);
    }
  }
#pragma unroll
  for (int g = 0; g < 4; ++g)
#pragma unroll
    for (int r = 0; r < 8; ++r) {
      int qi = qt * 16 + r + 8 * hi;
      ob[((size_t)b * T + qi) * C + hh * HD + g * 16 + ml] =
          (bf16_t)(acc[g][r] / lrow[r]);
    }
}

// ---------------- c_proj: out = resid + o@W^T + b  (writes d_out) --------------
__global__ void k_gemm_cproj(const bf16_t* __restrict__ A, const bf16_t* __restrict__ W,
                             const float* __restrict__ bias,
                             const float* __restrict__ resid,
                             float* __restrict__ out) {
  __shared__ bf16_t ldsw[64 * 32];
  int lane = threadIdx.x & 31, w = threadIdx.x >> 5;
  int nblk = blockIdx.x * 64;
  int m0   = blockIdx.y * 64 + w * 16;
  v8f acc[4] = {};
  gemm_core_blk(A, W, C, m0, nblk, ldsw, acc, lane);
  int ml = lane & 15, hi = (lane >> 4) & 1;
#pragma unroll
  for (int g = 0; g < 4; ++g)
#pragma unroll
    for (int r = 0; r < 8; ++r) {
      size_t row = m0 + r + 8 * hi;
      size_t col = nblk + g * 16 + ml;
      size_t off = row * C + col;
      out[off] = resid[off] + acc[g][r] + bias[col];
    }
}

// ---------------- router logits / noise logits --------------------------------
__global__ void k_router(const float* __restrict__ h, const float* __restrict__ rw,
                         const float* __restrict__ rb, const float* __restrict__ nw,
                         const float* __restrict__ nb, float* __restrict__ logits,
                         float* __restrict__ nlog) {
  int t = blockIdx.x;
  int lane = threadIdx.x & 31, wv = threadIdx.x >> 5; // 16 waves
  const float* w = (wv < EXP) ? (rw + (size_t)wv * C) : (nw + (size_t)(wv - EXP) * C);
  float s = 0.f;
  for (int c = lane; c < C; c += 32) s += h[(size_t)t * C + c] * w[c];
#pragma unroll
  for (int m = 16; m >= 1; m >>= 1) s += __shfl_xor(s, m, 32);
  if (lane == 0) {
    if (wv < EXP) logits[(size_t)t * EXP + wv] = s + rb[wv];
    else          nlog[(size_t)t * EXP + wv - EXP] = s + nb[wv - EXP];
  }
}

// ---------------- noisy top-2 gating ------------------------------------------
__global__ void k_topk(const float* __restrict__ logits, const float* __restrict__ nlog,
                       const float* __restrict__ noise, float* __restrict__ flat_g,
                       int* __restrict__ eid) {
  int t = blockIdx.x * blockDim.x + threadIdx.x;
  if (t >= N_TOK) return;
  float nz[EXP];
#pragma unroll
  for (int e = 0; e < EXP; ++e) {
    float xx = nlog[(size_t)t * EXP + e];
    float sp = (xx > 20.f) ? xx : log1pf(expf(xx));
    nz[e] = logits[(size_t)t * EXP + e] + noise[(size_t)t * EXP + e] * sp;
  }
  int i1 = 0; float v1 = nz[0];
#pragma unroll
  for (int e = 1; e < EXP; ++e) if (nz[e] > v1) { v1 = nz[e]; i1 = e; }
  int i2 = -1; float v2 = -INFINITY;
#pragma unroll
  for (int e = 0; e < EXP; ++e)
    if (e != i1 && nz[e] > v2) { v2 = nz[e]; i2 = e; }
  float e2 = expf(v2 - v1);
  float g1 = 1.f / (1.f + e2);
  float g2 = e2 / (1.f + e2);
  float row[EXP];
#pragma unroll
  for (int e = 0; e < EXP; ++e) row[e] = 0.f;
  row[i1] = g1;
  row[i2] = g2;
  int be = 0; float bv = row[0];
#pragma unroll
  for (int e = 1; e < EXP; ++e) if (row[e] > bv) { bv = row[e]; be = e; }
#pragma unroll
  for (int e = 0; e < EXP; ++e) flat_g[(size_t)t * EXP + e] = row[e];
  eid[t] = be;
}

// ---------------- capacity-limited assignment (ascending token order) ----------
__global__ void k_assign(const int* __restrict__ eid, const float* __restrict__ flat_g,
                         int* __restrict__ idx_e, float* __restrict__ gate_e) {
  int e = threadIdx.x;
  if (e >= EXP) return;
  int cnt = 0;
  for (int t = 0; t < N_TOK && cnt < CAP; ++t)
    if (eid[t] == e) {
      idx_e[e * CAP + cnt]  = t;
      gate_e[e * CAP + cnt] = flat_g[(size_t)t * EXP + e];
      ++cnt;
    }
  for (; cnt < CAP; ++cnt) {
    idx_e[e * CAP + cnt]  = N_TOK; // sentinel (invalid)
    gate_e[e * CAP + cnt] = 0.f;
  }
}

// gather expert inputs (safe index like reference)
__global__ void k_gather(const bf16_t* __restrict__ hb, const int* __restrict__ idx_e,
                         bf16_t* __restrict__ xe) {
  int i = blockIdx.x, e = blockIdx.y;
  int tok = idx_e[e * CAP + i];
  if (tok >= N_TOK) tok = N_TOK - 1;
  for (int c = threadIdx.x; c < C; c += 256)
    xe[((size_t)e * CAP + i) * C + c] = hb[(size_t)tok * C + c];
}

// ---------------- TDM prefetch of expert weights (gfx1250 tensor DMA) ----------
// Pulls 128x64-element bf16 tiles into LDS via the Tensor Data Mover, per the D#
// layout in CDNA5 ISA. Output-neutral: data is discarded; the
// effect is warming GL2 with the expert weights ahead of the FFN GEMMs.
__global__ void k_tdm_prefetch(const bf16_t* __restrict__ w, float* __restrict__ dummy,
                               long rows, long kdim) {
#if HAVE_TDM
  __shared__ bf16_t stage[128 * 64];     // 16 KB target for one tile
  long tile_r = ((long)blockIdx.x * 128) % rows;
  unsigned long long ga = (unsigned long long)(const bf16_t*)(w + tile_r * kdim);
  if (threadIdx.x == 0) {   // TDM issue ignores EXEC; one wave issues
    v4u g0;
    g0[0] = 1u;                                    // count=1 (valid user D#)
    g0[1] = 0u;                                    // lds_addr = 0 (wave LDS base)
    g0[2] = (unsigned)(ga & 0xFFFFFFFFull);        // global_addr[31:0]
    g0[3] = (unsigned)((ga >> 32) & 0x01FFFFFFull) // global_addr[56:32]
            | (2u << 30);                          // type = 2 ("image")
    v8i g1;
    g1[0] = (int)(1u << 16);                       // wg_mask=0, data_size=1 (2B)
    g1[1] = (int)((unsigned)(kdim & 0xFFFF) << 16);          // tensor_dim0 lo
    g1[2] = (int)(((unsigned)(kdim >> 16) & 0xFFFFu) |
                  ((unsigned)(rows & 0xFFFF) << 16));        // dim0 hi | dim1 lo
    g1[3] = (int)(((unsigned)(rows >> 16) & 0xFFFFu) |
                  (64u << 16));                              // dim1 hi | tile_dim0=64
    g1[4] = (int)128u;                             // tile_dim1=128, tile_dim2=0
    g1[5] = (int)(unsigned)kdim;                   // tensor_dim0_stride lo
    g1[6] = 0;                                     // stride hi | dim1_stride lo
    g1[7] = 0;                                     // dim1_stride hi
    v4i z4 = {0, 0, 0, 0};
    v8i z8 = {0, 0, 0, 0, 0, 0, 0, 0};
    // 6-arg form (clang-23 / therock headers): (g0, g1, g2, g3, g4, cpol)
    __builtin_amdgcn_tensor_load_to_lds(g0, g1, z4, z4, z8, 0);
    __builtin_amdgcn_s_wait_tensorcnt(0);
  }
  __syncthreads();
  float sv = (float)stage[threadIdx.x];
  if (sv == 12345.678f) dummy[threadIdx.x] = sv;   // keep LDS live; never taken
#else
  (void)w; (void)dummy; (void)rows; (void)kdim;
#endif
}

// ---------------- expert up-proj: h1 = relu(x@W1^T + b1)^2 (bf16) --------------
__global__ void k_gemm_exp1(const bf16_t* __restrict__ Xe, const bf16_t* __restrict__ W1,
                            const float* __restrict__ B1, bf16_t* __restrict__ H1) {
  __shared__ bf16_t ldsw[64 * 32];
  int lane = threadIdx.x & 31, w = threadIdx.x >> 5;
  int e    = blockIdx.z;
  int nblk = blockIdx.x * 64;
  int m0   = blockIdx.y * 64 + w * 16;
  const bf16_t* A  = Xe + (size_t)e * CAP * C;
  const bf16_t* Wp = W1 + (size_t)e * C4 * C;
  const float*  bp = B1 + (size_t)e * C4;
  bf16_t* op = H1 + (size_t)e * CAP * C4;
  v8f acc[4] = {};
  gemm_core_blk(A, Wp, C, m0, nblk, ldsw, acc, lane);
  int ml = lane & 15, hi = (lane >> 4) & 1;
#pragma unroll
  for (int g = 0; g < 4; ++g)
#pragma unroll
    for (int r = 0; r < 8; ++r) {
      int n  = nblk + g * 16 + ml;
      float v = acc[g][r] + bp[n];
      v = fmaxf(v, 0.f);
      op[(size_t)(m0 + r + 8 * hi) * C4 + n] = (bf16_t)(v * v);
    }
}

// ---------------- expert down-proj + gated scatter-add into d_out --------------
__global__ void k_gemm_exp2(const bf16_t* __restrict__ H1, const bf16_t* __restrict__ W2,
                            const float* __restrict__ B2, const int* __restrict__ idx_e,
                            const float* __restrict__ gate_e, float* __restrict__ out) {
  __shared__ bf16_t ldsw[64 * 32];
  int lane = threadIdx.x & 31, w = threadIdx.x >> 5;
  int e    = blockIdx.z;
  int nblk = blockIdx.x * 64;
  int m0   = blockIdx.y * 64 + w * 16;
  const bf16_t* A  = H1 + (size_t)e * CAP * C4;
  const bf16_t* Wp = W2 + (size_t)e * C * C4;
  const float*  bp = B2 + (size_t)e * C;
  const int*    ip = idx_e + (size_t)e * CAP;
  const float*  gp = gate_e + (size_t)e * CAP;
  v8f acc[4] = {};
  gemm_core_blk(A, Wp, C4, m0, nblk, ldsw, acc, lane);
  int ml = lane & 15, hi = (lane >> 4) & 1;
#pragma unroll
  for (int g = 0; g < 4; ++g)
#pragma unroll
    for (int r = 0; r < 8; ++r) {
      int row = m0 + r + 8 * hi;
      int tok = ip[row];
      if (tok < N_TOK) {  // each token owned by exactly one expert slot: no atomics
        int n = nblk + g * 16 + ml;
        size_t off = (size_t)tok * C + n;
        out[off] += (acc[g][r] + bp[n]) * gp[row];
      }
    }
}

// ---------------- host orchestration ------------------------------------------
extern "C" void kernel_launch(void* const* d_in, const int* in_sizes, int n_in,
                              void* d_out, int out_size, void* d_ws, size_t ws_size,
                              hipStream_t stream) {
  (void)in_sizes; (void)n_in; (void)out_size; (void)ws_size;
  const float* x        = (const float*)d_in[0];
  const float* x0       = (const float*)d_in[1];
  const float* noise    = (const float*)d_in[2];
  const float* lambdas  = (const float*)d_in[3];
  const float* lamb     = (const float*)d_in[4];
  const float* qkv_w    = (const float*)d_in[5];
  const float* c_proj_w = (const float*)d_in[6];
  const float* c_proj_b = (const float*)d_in[7];
  const float* router_w = (const float*)d_in[8];
  const float* router_b = (const float*)d_in[9];
  const float* noise_w  = (const float*)d_in[10];
  const float* noise_b  = (const float*)d_in[11];
  const float* ew1      = (const float*)d_in[12];
  const float* eb1      = (const float*)d_in[13];
  const float* ew2      = (const float*)d_in[14];
  const float* eb2      = (const float*)d_in[15];
  float* out = (float*)d_out;

  char* ws = (char*)d_ws;
  size_t off = 0;
  auto alloc = [&](size_t bytes) -> void* {
    off = (off + 255) & ~(size_t)255;
    void* p = ws + off;
    off += bytes;
    return p;
  };

  float*  x_res = (float*)alloc((size_t)N_TOK * C * 4);
  bf16_t* a_bf  = (bf16_t*)alloc((size_t)N_TOK * C * 2);
  bf16_t* wqkv  = (bf16_t*)alloc((size_t)3 * C * C * 2);
  bf16_t* wcp   = (bf16_t*)alloc((size_t)C * C * 2);
  bf16_t* wew1  = (bf16_t*)alloc((size_t)EXP * C4 * C * 2);
  bf16_t* wew2  = (bf16_t*)alloc((size_t)EXP * C * C4 * 2);
  size_t qkv_byte_off = (off + 255) & ~(size_t)255;   // h1 aliases here later
  float*  qkvf  = (float*)alloc((size_t)3 * N_TOK * C * 4);
  bf16_t* qbf   = (bf16_t*)alloc((size_t)Bc * H * T * HD * 2);
  bf16_t* kbf   = (bf16_t*)alloc((size_t)Bc * H * T * HD * 2);
  bf16_t* vtb   = (bf16_t*)alloc((size_t)Bc * H * T * HD * 2);
  bf16_t* obf   = (bf16_t*)alloc((size_t)N_TOK * C * 2);
  float*  h_f   = (float*)alloc((size_t)N_TOK * C * 4);
  bf16_t* h_bf  = (bf16_t*)alloc((size_t)N_TOK * C * 2);
  float*  logits = (float*)alloc((size_t)N_TOK * EXP * 4);
  float*  nlogit = (float*)alloc((size_t)N_TOK * EXP * 4);
  float*  flat_g = (float*)alloc((size_t)N_TOK * EXP * 4);
  int*    eid    = (int*)alloc((size_t)N_TOK * 4);
  int*    idx_e  = (int*)alloc((size_t)EXP * CAP * 4);
  float*  gate_e = (float*)alloc((size_t)EXP * CAP * 4);
  bf16_t* xe     = (bf16_t*)alloc((size_t)EXP * CAP * C * 2);
  float*  dummyf = (float*)alloc(512);
  // h1 (64 MB) aliases qkvf + qbf + kbf (dead after attention)
  bf16_t* h1 = (bf16_t*)(ws + qkv_byte_off);

  // 1) fp32 -> bf16 weight conversion (vectorized)
  k_cvt_bf16<<<512, 256, 0, stream>>>(qkv_w, wqkv, (long)3 * C * C);
  k_cvt_bf16<<<512, 256, 0, stream>>>(c_proj_w, wcp, (long)C * C);
  k_cvt_bf16<<<2048, 256, 0, stream>>>(ew1, wew1, (long)EXP * C4 * C);
  k_cvt_bf16<<<2048, 256, 0, stream>>>(ew2, wew2, (long)EXP * C * C4);

  // 2) residual mix + rmsnorm
  k_mix_rms<<<N_TOK, 256, 0, stream>>>(x, x0, lambdas, x_res, a_bf);

  // 3) QKV projection (WMMA bf16, LDS-staged W tiles)
  k_gemm_qkv<<<dim3(C / 64, N_TOK / 64, 3), 128, 0, stream>>>(a_bf, wqkv, qkvf);

  // 4) head rmsnorm + rotary + V transpose
  k_qk_prep<<<dim3(T, H, Bc), HD, 0, stream>>>(qkvf, lamb, qbf, kbf, vtb);

  // 5) causal flash attention (WMMA)
  k_attn<<<dim3(T / 16, H, Bc), 32, 0, stream>>>(qbf, kbf, vtb, obf);

  // 6) output projection + residual -> d_out
  k_gemm_cproj<<<dim3(C / 64, N_TOK / 64), 128, 0, stream>>>(obf, wcp, c_proj_b,
                                                             x_res, out);

  // 7) rmsnorm of post-attention state
  k_rms_h<<<N_TOK, 256, 0, stream>>>(out, h_f, h_bf);

  // 8) router + noise logits
  k_router<<<N_TOK, 512, 0, stream>>>(h_f, router_w, router_b, noise_w, noise_b,
                                      logits, nlogit);

  // 9) noisy top-2 gating
  k_topk<<<N_TOK / 256, 256, 0, stream>>>(logits, nlogit, noise, flat_g, eid);

  // 10) capacity assignment + gather
  k_assign<<<1, 32, 0, stream>>>(eid, flat_g, idx_e, gate_e);
  k_gather<<<dim3(CAP, EXP), 256, 0, stream>>>(h_bf, idx_e, xe);

  // 11) TDM prefetch of expert weights into GL2 (gfx1250 tensor DMA path)
  k_tdm_prefetch<<<64, 128, 0, stream>>>(wew1, dummyf, (long)EXP * C4, (long)C);
  k_tdm_prefetch<<<64, 128, 0, stream>>>(wew2, dummyf, (long)EXP * C, (long)C4);

  // 12) expert FFN (WMMA) + gated scatter-add
  k_gemm_exp1<<<dim3(C4 / 64, CAP / 64, EXP), 128, 0, stream>>>(xe, wew1, eb1, h1);
  k_gemm_exp2<<<dim3(C / 64, CAP / 64, EXP), 128, 0, stream>>>(h1, wew2, eb2, idx_e,
                                                               gate_e, out);
}
